// Attention_14293651161426
// MI455X (gfx1250) — compile-verified, matrix operand order checked by reading, not run on hardware
//
#include <hip/hip_runtime.h>
#include <hip/hip_bf16.h>
#include <stdint.h>

#define S_LEN 2048
#define HID_  2048
#define NH_   16
#define NKV_  4
#define HD_   128

typedef __attribute__((ext_vector_type(16))) __bf16 v16bf;
typedef __attribute__((ext_vector_type(8)))  __bf16 v8bf;
typedef __attribute__((ext_vector_type(8)))  float  v8f;

// ---------- helpers ----------

__device__ __forceinline__ __bf16 f2bf(float f) {
  unsigned u = __builtin_bit_cast(unsigned, f);
  unsigned r = (u + 0x7FFFu + ((u >> 16) & 1u)) >> 16;
  unsigned short h = (unsigned short)r;
  return __builtin_bit_cast(__bf16, h);
}

// Load one 16-element bf16 fragment for a 16x32 WMMA operand.
// CDNA5 16-bit A/B layout: lane half (lane>=16) selects K offset +8,
// elements 0..7 -> K = half8 + 0..7, elements 8..15 -> K = 16 + half8 + 0..7.
__device__ __forceinline__ v16bf load_frag(const __bf16* p, int half8) {
  v8bf lo = *reinterpret_cast<const v8bf*>(p + half8);
  v8bf hi = *reinterpret_cast<const v8bf*>(p + half8 + 16);
  v16bf r;
#pragma unroll
  for (int i = 0; i < 8; ++i) { r[i] = lo[i]; r[i + 8] = hi[i]; }
  return r;
}

__device__ __forceinline__ v8f wmma_bf16(v16bf a, v16bf b, v8f c) {
  return __builtin_amdgcn_wmma_f32_16x16x32_bf16(
      /*neg_a=*/false, a, /*neg_b=*/false, b,
      /*c_mod=*/(short)0, c, /*reuse_a=*/false, /*reuse_b=*/false);
}

// ---------- async global -> LDS copy (ASYNCcnt path) ----------

typedef int v4i_ __attribute__((vector_size(16)));
typedef __attribute__((address_space(1))) v4i_ global_v4i;
typedef __attribute__((address_space(3))) v4i_ lds_v4i;

__device__ __forceinline__ void async_copy_b128(const void* gsrc, void* ldst) {
#if __has_builtin(__builtin_amdgcn_global_load_async_to_lds_b128)
  __builtin_amdgcn_global_load_async_to_lds_b128(
      (global_v4i*)const_cast<void*>(gsrc),
      (lds_v4i*)ldst, 0, 0);
#else
  asm volatile("global_load_async_to_lds_b128 %0, %1, off"
               :
               : "v"((unsigned)(size_t)(__attribute__((address_space(3))) void*)ldst),
                 "v"(gsrc)
               : "memory");
#endif
}

#if __has_builtin(__builtin_amdgcn_s_wait_asynccnt)
#define WAIT_ASYNC(n) __builtin_amdgcn_s_wait_asynccnt(n)
#else
#define WAIT_ASYNC(n) asm volatile("s_wait_asynccnt %0" ::"n"(n) : "memory")
#endif

// ---------- fp32 -> bf16 conversion ----------

__global__ void bf16_conv(const float* __restrict__ in, __bf16* __restrict__ out, int n) {
  int i = blockIdx.x * blockDim.x + threadIdx.x;
  if (i < n) out[i] = f2bf(in[i]);
}

// ---------- generic GEMM: C[m][n] = sum_k A[m][k] * B[n][k] ----------
// A: M x K bf16 row-major, B: N x K bf16 row-major, C: M x N fp32 row-major.
// Block = 256 threads (8 waves), tile M=128 x N=64, K-slab 64 (two 32-chunks).
// Double-buffered async staging of A(128x64) and B(64x64) slabs into LDS.
// Per slab: 6 async b128/thread, s_wait_asynccnt 6 retires the previous slab
// while WMMAs run; fragments preloaded so WMMAs issue back-to-back.
// grid = (N/64, M/128). Requires K % 64 == 0.

#define GEMM_M 128
#define GEMM_N 64
#define KS     64

__global__ __launch_bounds__(256) void gemm_bf16_nt(const __bf16* __restrict__ A,
                                                    const __bf16* __restrict__ B,
                                                    float* __restrict__ C,
                                                    int N, int K) {
  __shared__ __align__(16) __bf16 ldsA[2][GEMM_M * KS];   // 2 x 16 KiB
  __shared__ __align__(16) __bf16 ldsB[2][GEMM_N * KS];   // 2 x 8 KiB

  const int tid   = threadIdx.x;
  const int wave  = tid >> 5;   // wave32
  const int lane  = tid & 31;
  const int m16   = lane & 15;
  const int half8 = (lane & 16) ? 8 : 0;
  const int rblk  = blockIdx.y * GEMM_M;
  const int cblk  = blockIdx.x * GEMM_N;

  // cooperative load slots: A -> 4 x b128 per thread, B -> 2 x b128 per thread
  const int aRow = tid >> 1;            // 0..127
  const int aCol = (tid & 1) * 32;      // elements (four adjacent b128)
  const int bRow = tid >> 2;            // 0..63
  const int bCol = (tid & 3) * 16;      // elements (two adjacent b128)

  const __bf16* gA = A + (size_t)(rblk + aRow) * K + aCol;
  const __bf16* gB = B + (size_t)(cblk + bRow) * K + bCol;
  __bf16* lA = &ldsA[0][aRow * KS + aCol];
  __bf16* lB = &ldsB[0][bRow * KS + bCol];

  auto issue = [&](int kk, int buf) {
    __bf16* la = lA + buf * (GEMM_M * KS);
    const __bf16* ga = gA + kk;
    async_copy_b128(ga,      la);
    async_copy_b128(ga + 8,  la + 8);
    async_copy_b128(ga + 16, la + 16);
    async_copy_b128(ga + 24, la + 24);
    __bf16* lb = lB + buf * (GEMM_N * KS);
    const __bf16* gb = gB + kk;
    async_copy_b128(gb,     lb);
    async_copy_b128(gb + 8, lb + 8);
  };

  v8f acc[4] = {};
  const int steps = K / KS;
  issue(0, 0);
  for (int i = 0; i < steps; ++i) {
    const int cur = i & 1;
    if (i + 1 < steps) {
      issue((i + 1) * KS, cur ^ 1);   // prefetch next slab into other buffer
      WAIT_ASYNC(6);                  // retire previous slab (in-order)
    } else {
      WAIT_ASYNC(0);
    }
    __syncthreads();                  // all waves' slab writes visible

#pragma unroll
    for (int kc = 0; kc < 2; ++kc) {
      v16bf af = load_frag(&ldsA[cur][(wave * 16 + m16) * KS + kc * 32], half8);
      v16bf bfr[4];
#pragma unroll
      for (int t = 0; t < 4; ++t)
        bfr[t] = load_frag(&ldsB[cur][(t * 16 + m16) * KS + kc * 32], half8);
#pragma unroll
      for (int t = 0; t < 4; ++t)
        acc[t] = wmma_bf16(af, bfr[t], acc[t]);
    }
    __syncthreads();                  // done reading before buffer reuse
  }

#pragma unroll
  for (int t = 0; t < 4; ++t) {
    const int col = cblk + t * 16 + m16;
#pragma unroll
    for (int j = 0; j < 8; ++j) {
      C[(size_t)(rblk + wave * 16 + half8 + j) * N + col] = acc[t][j];
    }
  }
}

// ---------- RoPE + head-major repack ----------
// in:  fp32 [S][nheads*HD] (GEMM output), cos/sin fp32 [S][HD] (first 64 used)
// out: bf16 [nheads][S][HD]

__global__ void rope_pack(const float* __restrict__ inp, const float* __restrict__ cosb,
                          const float* __restrict__ sinb, __bf16* __restrict__ outp,
                          int nheads) {
  int idx = blockIdx.x * blockDim.x + threadIdx.x;     // (s*nheads + h)*64 + i
  if (idx >= S_LEN * nheads * 64) return;
  int i  = idx & 63;
  int sh = idx >> 6;
  int h  = sh % nheads;
  int s  = sh / nheads;
  const float* v = inp + ((size_t)s * nheads + h) * HD_;
  float y  = (i & 1) ? v[i - 1] : -v[i + 1];           // rotate-half pairs
  float xp = v[64 + i];
  float cs = cosb[(size_t)s * HD_ + i];
  float sn = sinb[(size_t)s * HD_ + i];
  __bf16* o = outp + ((size_t)h * S_LEN + s) * HD_;
  o[i]      = f2bf(y * cs + xp * sn);
  o[64 + i] = f2bf(-y * sn + xp * cs);
}

// ---------- V repack (transposed): fp32 [S][NKV*HD] -> bf16 [NKV][HD][S] ----------

__global__ void pack_v(const float* __restrict__ inp, __bf16* __restrict__ outp) {
  int idx = blockIdx.x * blockDim.x + threadIdx.x;     // s*512 + kv*128 + d
  if (idx >= S_LEN * NKV_ * HD_) return;
  int s  = idx >> 9;
  int r  = idx & 511;
  int kv = r >> 7;
  int d  = r & 127;
  outp[((size_t)kv * HD_ + d) * S_LEN + s] = f2bf(inp[idx]);
}

// ---------- fused causal attention (GQA) ----------
// One workgroup = (head h, 16-query tile qt). 320 KB WGP LDS lets us keep the
// whole 16 x L score strip (fp32, 128 KB) + bf16 probabilities (64 KB) resident.

#define SC_PITCH 2048
#define PB_PITCH 2056   // multiple of 8 -> 16B aligned bf16 rows

__global__ __launch_bounds__(256) void attn_fused(const __bf16* __restrict__ qb,
                                                  const __bf16* __restrict__ kb,
                                                  const __bf16* __restrict__ vtb,
                                                  __bf16* __restrict__ ob) {
  extern __shared__ char smem[];
  float*  sc = (float*)smem;                                  // [16][SC_PITCH]
  __bf16* pb = (__bf16*)(smem + (size_t)16 * SC_PITCH * 4);   // [16][PB_PITCH]
  __shared__ float red[16][16];
  __shared__ float rowred[16];

  const int qt    = blockIdx.x;
  const int h     = blockIdx.y;
  const int kvh   = h >> 2;               // NH/NKV = 4
  const int qbase = qt * 16;
  const int tid   = threadIdx.x;
  const int wave  = tid >> 5;
  const int lane  = tid & 31;
  const int m16   = lane & 15;
  const int half8 = (lane & 16) ? 8 : 0;
  const int ntiles = qt + 1;              // causal: k-tiles 0..qt
  const int L  = ntiles * 16;
  const int Lp = (L + 31) & ~31;
  const float scale = 0.08838834764831845f;   // 1/sqrt(128)

  // ---- Phase 1: S = Q K^T (scaled, causal-masked) into LDS ----
  v16bf qf[4];
  {
    const __bf16* qrow = qb + ((size_t)h * S_LEN + qbase + m16) * HD_;
#pragma unroll
    for (int c = 0; c < 4; ++c) qf[c] = load_frag(qrow + c * 32, half8);
  }
  {
    int kt = wave;
    v16bf kf[4];
    if (kt < ntiles) {
      const __bf16* krow = kb + ((size_t)kvh * S_LEN + kt * 16 + m16) * HD_;
#pragma unroll
      for (int cc = 0; cc < 4; ++cc) kf[cc] = load_frag(krow + cc * 32, half8);
    }
    for (; kt < ntiles; kt += 8) {
      const int ktn = kt + 8;
      v16bf kn[4];
      if (ktn < ntiles) {            // prefetch next K tile during WMMAs
        const __bf16* krow = kb + ((size_t)kvh * S_LEN + ktn * 16 + m16) * HD_;
#pragma unroll
        for (int cc = 0; cc < 4; ++cc) kn[cc] = load_frag(krow + cc * 32, half8);
      }
      v8f c = {};
#pragma unroll
      for (int cc = 0; cc < 4; ++cc) c = wmma_bf16(qf[cc], kf[cc], c);
      const int kcol = kt * 16 + m16;
#pragma unroll
      for (int j = 0; j < 8; ++j) {
        const int row = half8 + j;
        float v = (kcol <= qbase + row) ? c[j] * scale : -3.0e38f;
        sc[row * SC_PITCH + kcol] = v;
      }
      if (ktn < ntiles) {
#pragma unroll
        for (int cc = 0; cc < 4; ++cc) kf[cc] = kn[cc];
      }
    }
  }
  __syncthreads();

  // ---- Phase 2: row softmax over L keys, write bf16 P (zero-padded to Lp) ----
  {
    const int r   = tid >> 4;
    const int c16 = tid & 15;
    float mx = -3.4e38f;
    for (int c = c16; c < L; c += 16) mx = fmaxf(mx, sc[r * SC_PITCH + c]);
    red[r][c16] = mx;
    __syncthreads();
    if (c16 == 0) {
      float m2 = red[r][0];
#pragma unroll
      for (int i = 1; i < 16; ++i) m2 = fmaxf(m2, red[r][i]);
      rowred[r] = m2;
    }
    __syncthreads();
    const float rm = rowred[r];
    float sum = 0.f;
    for (int c = c16; c < L; c += 16) {
      float e = __expf(sc[r * SC_PITCH + c] - rm);
      sc[r * SC_PITCH + c] = e;
      sum += e;
    }
    red[r][c16] = sum;
    __syncthreads();
    if (c16 == 0) {
      float s2 = 0.f;
#pragma unroll
      for (int i = 0; i < 16; ++i) s2 += red[r][i];
      rowred[r] = s2;
    }
    __syncthreads();
    const float inv = 1.0f / rowred[r];
    for (int c = c16; c < L; c += 16)
      pb[r * PB_PITCH + c] = f2bf(sc[r * SC_PITCH + c] * inv);
    for (int c = L + c16; c < Lp; c += 16)
      pb[r * PB_PITCH + c] = f2bf(0.0f);
  }
  __syncthreads();

  // ---- Phase 3: O = P @ V  (wave w -> 16-wide d slice), reg double-buffered ----
  {
    const int dbase = wave * 16;
    const __bf16* vrow = vtb + ((size_t)kvh * HD_ + dbase + m16) * S_LEN;
    const __bf16* prow = pb + m16 * PB_PITCH;
    v8f o = {};
    v16bf a0 = load_frag(prow, half8);
    v16bf b0 = load_frag(vrow, half8);
    for (int lb = 0; lb < Lp; lb += 32) {
      v16bf a1 = a0, b1 = b0;
      if (lb + 32 < Lp) {            // prefetch next chunk during WMMA
        a1 = load_frag(prow + lb + 32, half8);
        b1 = load_frag(vrow + lb + 32, half8);
      }
      o = wmma_bf16(a0, b0, o);
      a0 = a1; b0 = b1;
    }
#pragma unroll
    for (int j = 0; j < 8; ++j) {
      ob[(size_t)(qbase + half8 + j) * HID_ + (size_t)h * HD_ + dbase + m16] = f2bf(o[j]);
    }
  }
}

// ---------- host launch ----------

extern "C" void kernel_launch(void* const* d_in, const int* in_sizes, int n_in,
                              void* d_out, int out_size, void* d_ws, size_t ws_size,
                              hipStream_t stream) {
  (void)in_sizes; (void)n_in; (void)out_size; (void)ws_size;
  const float* x    = (const float*)d_in[0];
  const float* cosb = (const float*)d_in[1];
  const float* sinb = (const float*)d_in[2];
  const float* wq   = (const float*)d_in[3];
  const float* wk   = (const float*)d_in[4];
  const float* wv   = (const float*)d_in[5];
  const float* wo   = (const float*)d_in[6];
  float* out = (float*)d_out;

  char* ws = (char*)d_ws;
  // workspace layout (bytes), total 64 MiB
  __bf16* xb  = (__bf16*)(ws + 0);                 // 8 MiB  : x bf16 [S][HID]
  __bf16* wqb = (__bf16*)(ws + 8388608);           // 8 MiB
  __bf16* wkb = (__bf16*)(ws + 16777216);          // 2 MiB
  __bf16* wvb = (__bf16*)(ws + 18874368);          // 2 MiB
  __bf16* wob = (__bf16*)(ws + 20971520);          // 8 MiB
  float*  Qf  = (float*)(ws + 29360128);           // 16 MiB : [S][2048] fp32
  float*  Kf  = (float*)(ws + 46137344);           // 4 MiB  : [S][512]  fp32
  float*  Vf  = (float*)(ws + 50331648);           // 4 MiB  : [S][512]  fp32
  __bf16* qhb = (__bf16*)(ws + 54525952);          // 8 MiB  : [NH][S][HD]
  __bf16* khb = (__bf16*)(ws + 62914560);          // 2 MiB  : [NKV][S][HD]
  __bf16* vtb = (__bf16*)(ws + 65011712);          // 2 MiB  : [NKV][HD][S]
  __bf16* obf = (__bf16*)(ws + 29360128);          // reuse Qf region (dead by then)

  const int nx = S_LEN * HID_;          // 4194304
  const int nk = 512 * HID_;            // 1048576

  // 1) convert inputs/weights to bf16
  bf16_conv<<<nx / 256, 256, 0, stream>>>(x,  xb,  nx);
  bf16_conv<<<nx / 256, 256, 0, stream>>>(wq, wqb, nx);
  bf16_conv<<<nk / 256, 256, 0, stream>>>(wk, wkb, nk);
  bf16_conv<<<nk / 256, 256, 0, stream>>>(wv, wvb, nk);
  bf16_conv<<<nx / 256, 256, 0, stream>>>(wo, wob, nx);

  // 2) projections: Qf = x wq^T, Kf = x wk^T, Vf = x wv^T
  gemm_bf16_nt<<<dim3(2048 / 64, S_LEN / 128), 256, 0, stream>>>(xb, wqb, Qf, 2048, HID_);
  gemm_bf16_nt<<<dim3(512  / 64, S_LEN / 128), 256, 0, stream>>>(xb, wkb, Kf, 512,  HID_);
  gemm_bf16_nt<<<dim3(512  / 64, S_LEN / 128), 256, 0, stream>>>(xb, wvb, Vf, 512,  HID_);

  // 3) RoPE + repack
  rope_pack<<<(S_LEN * NH_  * 64) / 256, 256, 0, stream>>>(Qf, cosb, sinb, qhb, NH_);
  rope_pack<<<(S_LEN * NKV_ * 64) / 256, 256, 0, stream>>>(Kf, cosb, sinb, khb, NKV_);
  pack_v<<<(S_LEN * NKV_ * HD_) / 256, 256, 0, stream>>>(Vf, vtb);

  // 4) fused causal attention -> obf bf16 [S][NH*HD]
  const size_t smem_bytes = (size_t)16 * SC_PITCH * 4 + (size_t)16 * PB_PITCH * 2; // 196864
  attn_fused<<<dim3(S_LEN / 16, NH_), 256, smem_bytes, stream>>>(qhb, khb, vtb, obf);

  // 5) output projection: out = obf wo^T
  gemm_bf16_nt<<<dim3(2048 / 64, S_LEN / 128), 256, 0, stream>>>(obf, wob, out, 2048, HID_);
}
